// ExpertGraphConv_63213328662859
// MI455X (gfx1250) — matrix operand order repaired and checked by reading, not compiled, not run on hardware
//
#include <hip/hip_runtime.h>
#include <math.h>

// ---------------------------------------------------------------------------
// ExpertGraphConv fused kernel for gfx1250 (MI455X), wave32 + WMMA bf16.
//   B=8, S=512, E=16, D=512  ->  4096 tokens, each a 16x512 block.
// Heavy GEMMs (y@Wn^T, x@Ws^T) run on v_wmma_f32_16x16x32_bf16.
// TPB=4 tokens per workgroup: each weight B-fragment feeds 4 WMMAs,
// cutting L2 weight traffic to ~1 GB total (below WMMA compute time).
// GEMM runs as two matrix passes (Y@Wn^T, then X@Ws^T into the same
// accumulators) to keep live A-fragment registers at 32 instead of 64.
// ---------------------------------------------------------------------------

typedef __attribute__((ext_vector_type(16))) __bf16 v16bf;
typedef __attribute__((ext_vector_type(8)))  float  v8f;

constexpr int Em  = 16;      // experts (M dim of per-token GEMM)
constexpr int Dm  = 512;     // feature dim (K and N of GEMM)
constexpr int KS  = 16;      // K steps: Dm / 32
constexpr int NT  = 32;      // N tiles:  Dm / 16
constexpr int DP  = Dm + 8;  // padded LDS row (halves): 520*2B=1040B -> bank-offset 4/row
constexpr int TPB = 4;       // tokens per block

__device__ __forceinline__ unsigned short f32_to_bf16(float f) {
  unsigned int u = __float_as_uint(f);
  unsigned int r = u + 0x7FFFu + ((u >> 16) & 1u);   // round-to-nearest-even
  return (unsigned short)(r >> 16);
}
__device__ __forceinline__ float bf16_to_f32(unsigned short h) {
  return __uint_as_float(((unsigned int)h) << 16);
}

union FragBF {
  v16bf v;
  uint4 q[2];
};

// ---------------------------------------------------------------------------
// Kernel 1: pack Wn / Ws (f32, row-major [n][k]) into bf16 B-fragment order.
// Layout: packed[ mat(2) ][ nt(32) ][ ks(16) ][ lane(32) ][ e(16) ]
//   B[k][n] = W[n][k] ; lane L -> n = nt*16 + (L&15),
//   k = ks*32 + (L>=16 ? 16 : 0) + e   (16 contiguous k per lane -> b128 loads)
// ---------------------------------------------------------------------------
__global__ void pack_weights_k(const float* __restrict__ Wn,
                               const float* __restrict__ Ws,
                               unsigned short* __restrict__ packed) {
  const int total = 2 * NT * KS * 32 * 16;  // 524288
  for (int idx = blockIdx.x * blockDim.x + threadIdx.x; idx < total;
       idx += gridDim.x * blockDim.x) {
    int e    = idx & 15;
    int lane = (idx >> 4) & 31;
    int ks   = (idx >> 9) & 15;
    int nt   = (idx >> 13) & 31;
    int mat  = (idx >> 18) & 1;
    int n = nt * 16 + (lane & 15);
    int k = ks * 32 + ((lane >> 4) << 4) + e;
    const float* W = mat ? Ws : Wn;
    packed[idx] = f32_to_bf16(W[n * Dm + k]);
  }
}

// ---------------------------------------------------------------------------
// Kernel 2: one workgroup (8 waves) per TPB tokens.
// ---------------------------------------------------------------------------
__global__ __launch_bounds__(256)
void expert_graph_conv_k(const float* __restrict__ xg,           // [ntok,16,512]
                         const unsigned short* __restrict__ wpk, // packed weights
                         const float* __restrict__ bn,
                         const float* __restrict__ bs,
                         const float* __restrict__ mw_w,         // [1024]
                         const float* __restrict__ mw_b,         // [1]
                         const float* __restrict__ adj,          // [16,16]
                         float* __restrict__ out)                // [ntok,16,512]
{
  __shared__ __align__(16) unsigned short xbf[TPB][Em][DP];  // 65 KB
  __shared__ __align__(16) unsigned short ybf[TPB][Em][DP];  // 65 KB
  __shared__ float gate[TPB][Em][Em];                        // 4 KB
  __shared__ float red[TPB][2][Em][16];                      // 8 KB
  __shared__ float ac[TPB][2][Em];

  const int tid  = threadIdx.x;
  const int tok0 = blockIdx.x * TPB;

  // ---- Phase A0: load x (coalesced float4), convert to bf16 in LDS --------
  {
    const float4* src =
        reinterpret_cast<const float4*>(xg + (size_t)tok0 * Em * Dm);
    for (int i = tid; i < (TPB * Em * Dm) / 4; i += 256) {
      float4 v = src[i];
      int flat = i * 4;
      int t = flat >> 13;              // /(Em*Dm)
      int e = (flat >> 9) & 15;
      int d = flat & (Dm - 1);
      unsigned short* row = &xbf[t][e][d];
      row[0] = f32_to_bf16(v.x);
      row[1] = f32_to_bf16(v.y);
      row[2] = f32_to_bf16(v.z);
      row[3] = f32_to_bf16(v.w);
    }
  }
  __syncthreads();

  // ---- Phase A1: a = x@w1, c = x@w2 (16 partial lanes per expert) ---------
  {
    int e = tid >> 4, s = tid & 15;
    for (int t = 0; t < TPB; t++) {
      float pa = 0.f, pc = 0.f;
      for (int d = s; d < Dm; d += 16) {
        float xv = bf16_to_f32(xbf[t][e][d]);
        pa = fmaf(xv, mw_w[d], pa);
        pc = fmaf(xv, mw_w[Dm + d], pc);
      }
      red[t][0][e][s] = pa;
      red[t][1][e][s] = pc;
    }
  }
  __syncthreads();
  if (tid < TPB * 2 * Em) {
    int t = tid >> 5, v = (tid >> 4) & 1, e = tid & 15;
    float sum = 0.f;
    #pragma unroll
    for (int j = 0; j < 16; j++) sum += red[t][v][e][j];
    ac[t][v][e] = sum;
  }
  __syncthreads();

  // ---- Phase A2: gate[i][j] = sig(adj) * sig(a_i + c_j + b), zero diag ----
  {
    int i = tid >> 4, j = tid & 15;
    float adjv = 1.f / (1.f + expf(-adj[i * Em + j]));
    float mb = mw_b[0];
    for (int t = 0; t < TPB; t++) {
      float lg = ac[t][0][i] + ac[t][1][j] + mb;
      float g = adjv / (1.f + expf(-lg));
      gate[t][i][j] = (i == j) ? 0.f : g;
    }
  }
  __syncthreads();

  // ---- Phase A3: y = gate @ x  (f32 accumulate, store bf16) ---------------
  for (int t = 0; t < TPB; t++) {
    for (int idx = tid; idx < Em * Dm; idx += 256) {
      int e = idx >> 9, d = idx & (Dm - 1);
      float sum = 0.f;
      #pragma unroll
      for (int j = 0; j < Em; j++)
        sum = fmaf(gate[t][e][j], bf16_to_f32(xbf[t][j][d]), sum);
      ybf[t][e][d] = f32_to_bf16(sum);
    }
  }
  __syncthreads();

  // ---- Phase B: out = gelu( y @ Wn^T + x @ Ws^T + bn + bs ) via WMMA ------
  const int wave = tid >> 5;      // 0..7 -> N tiles [4w, 4w+3]
  const int lane = tid & 31;
  const int m    = lane & 15;     // A row / C column within tile
  const int hi   = lane >> 4;     // lane group

  v8f acc[TPB][4];
  #pragma unroll
  for (int t = 0; t < TPB; t++)
    #pragma unroll
    for (int q = 0; q < 4; q++)
      acc[t][q] = (v8f){0.f, 0.f, 0.f, 0.f, 0.f, 0.f, 0.f, 0.f};

  const size_t matStride = (size_t)NT * KS * 32 * 16;  // halves per matrix

  // Two passes: mat 0 -> A = y (ybf), B = Wn^T ; mat 1 -> A = x (xbf), B = Ws^T
  #pragma unroll
  for (int mat = 0; mat < 2; mat++) {
    const unsigned short(*src)[Em][DP] = mat ? xbf : ybf;
    const unsigned short* wbase = wpk + (size_t)mat * matStride;

    #pragma unroll 2
    for (int ks = 0; ks < KS; ks++) {
      // prefetch next K-step's weight fragments (global_prefetch_b8)
      if (ks + 1 < KS) {
        const unsigned short* pf =
            wbase + (((size_t)(wave * 4) * KS + (ks + 1)) * 32 + lane) * 16;
        __builtin_prefetch(pf, 0, 1);
      }

      // A fragments from LDS: lanes 0-15 hold K 0-15, lanes 16-31 hold K 16-31
      const int ka = ks * 32 + hi * 16;
      FragBF A[TPB];
      #pragma unroll
      for (int t = 0; t < TPB; t++) {
        A[t].q[0] = *(const uint4*)&src[t][m][ka];
        A[t].q[1] = *(const uint4*)&src[t][m][ka + 8];
      }

      #pragma unroll
      for (int q = 0; q < 4; q++) {
        const int nt = wave * 4 + q;
        const unsigned short* bp =
            wbase + (((size_t)nt * KS + ks) * 32 + lane) * 16;
        FragBF Bf;
        Bf.q[0] = *(const uint4*)bp;
        Bf.q[1] = *(const uint4*)(bp + 8);

        #pragma unroll
        for (int t = 0; t < TPB; t++) {
          acc[t][q] = __builtin_amdgcn_wmma_f32_16x16x32_bf16(
              false, A[t].v, false, Bf.v, (short)0, acc[t][q], false, false);
        }
      }
    }
  }

  // ---- Epilogue: bias + exact GELU, C layout: VGPR r -> row r + 8*hi ------
  #pragma unroll
  for (int q = 0; q < 4; q++) {
    const int n = (wave * 4 + q) * 16 + m;
    const float bias = bn[n] + bs[n];
    #pragma unroll
    for (int t = 0; t < TPB; t++) {
      #pragma unroll
      for (int r = 0; r < 8; r++) {
        const int row = r + hi * 8;
        float val = acc[t][q][r] + bias;
        float g = 0.5f * val * (1.f + erff(val * 0.70710678118654752f));
        out[((size_t)(tok0 + t) * Em + row) * Dm + n] = g;
      }
    }
  }
}

// ---------------------------------------------------------------------------
extern "C" void kernel_launch(void* const* d_in, const int* in_sizes, int n_in,
                              void* d_out, int out_size, void* d_ws, size_t ws_size,
                              hipStream_t stream) {
  const float* xg   = (const float*)d_in[0];  // expert_features
  const float* Wn   = (const float*)d_in[1];
  const float* bn   = (const float*)d_in[2];
  const float* Ws   = (const float*)d_in[3];
  const float* bs   = (const float*)d_in[4];
  const float* mw_w = (const float*)d_in[5];
  const float* mw_b = (const float*)d_in[6];
  const float* adj  = (const float*)d_in[7];
  float* out = (float*)d_out;

  unsigned short* wpk = (unsigned short*)d_ws;  // 2*512*512 bf16 = 1 MB

  const int ntok = in_sizes[0] / (Em * Dm);     // 4096

  pack_weights_k<<<512, 256, 0, stream>>>(Wn, Ws, wpk);
  expert_graph_conv_k<<<ntok / TPB, 256, 0, stream>>>(xg, wpk, bn, bs, mw_w,
                                                      mw_b, adj, out);
}
